// TopoEdgeModel_2293512536361
// MI455X (gfx1250) — matrix operand order repaired and checked by reading, not compiled, not run on hardware
//
#include <hip/hip_runtime.h>
#include <hip/hip_bf16.h>

typedef __attribute__((ext_vector_type(16))) _Float16 v16h;
typedef __attribute__((ext_vector_type(8)))  float    v8f;
typedef __attribute__((ext_vector_type(4)))  int      v4i;

#define BN  53
#define EE  2809
#define DD  256
#define HH  4
#define DSUB 64
#define BBATCH 4
#define LBB 64
#define MEROWS (BBATCH*EE)   // 11236
#define VTLD 2816            // padded key stride for transposed V (self-attn)

#if defined(__HIP_DEVICE_COMPILE__) &&                                        \
    __has_builtin(__builtin_amdgcn_global_load_async_to_lds_b128) &&          \
    __has_builtin(__builtin_amdgcn_s_wait_asynccnt)
#define HAVE_ASYNC_LDS 1
#endif

__device__ __forceinline__ unsigned laneid() { return threadIdx.x & 31u; }

// Stage 16 halfs (32B) from global to LDS: async DMA path on CDNA5 when the
// builtins exist, else a synchronous vector copy.
__device__ __forceinline__ void stage16(const _Float16* g, _Float16* l) {
#ifdef HAVE_ASYNC_LDS
  __builtin_amdgcn_global_load_async_to_lds_b128(
      (__attribute__((address_space(1))) v4i*)g,
      (__attribute__((address_space(3))) v4i*)l, 0, 0);
  __builtin_amdgcn_global_load_async_to_lds_b128(
      (__attribute__((address_space(1))) v4i*)(g + 8),
      (__attribute__((address_space(3))) v4i*)(l + 8), 0, 0);
#else
  const uint4* gp = reinterpret_cast<const uint4*>(g);
  uint4 v0 = gp[0], v1 = gp[1];
  uint4* lp = reinterpret_cast<uint4*>(l);
  lp[0] = v0; lp[1] = v1;
#endif
}

__device__ __forceinline__ void zfill16(_Float16* l) {
  uint4 z = {0, 0, 0, 0};
  uint4* lp = reinterpret_cast<uint4*>(l);
  lp[0] = z; lp[1] = z;
}

__device__ __forceinline__ void async_fence() {
#ifdef HAVE_ASYNC_LDS
  __builtin_amdgcn_s_wait_asynccnt(0);
#endif
}

__device__ __forceinline__ v8f zero8() {
  v8f z;
#pragma unroll
  for (int i = 0; i < 8; ++i) z[i] = 0.0f;
  return z;
}

// A-fragment (16x32 f16, MxK) from row-major LDS -> 2x ds_load_b128 per lane.
__device__ __forceinline__ v16h load_a_frag(const _Float16* base, int stride) {
  unsigned lane = laneid();
  unsigned m = lane & 15u, kh = lane >> 4;
  v16h a;
#pragma unroll
  for (int v = 0; v < 8; ++v) {
    int k = ((v >> 2) << 4) + ((int)kh << 3) + ((v & 3) << 1);
    const _Float16* p = base + (int)m * stride + k;
    a[2 * v] = p[0];
    a[2 * v + 1] = p[1];
  }
  return a;
}

// B-fragment (32x16, KxN) from a TRANSPOSED (N-major) LDS tile: per lane two
// 8-half runs at row n (k = 8*kh and 16 + 8*kh) -> 2x ds_load_b128.
__device__ __forceinline__ v16h load_bt_frag(const _Float16* base, int stride) {
  unsigned lane = laneid();
  unsigned n = lane & 15u, kh = lane >> 4;
  const _Float16* p = base + (int)n * stride + ((int)kh << 3);
  v16h b;
#pragma unroll
  for (int v = 0; v < 16; ++v) b[v] = p[((v >> 3) << 4) + (v & 7)];
  return b;
}

__device__ __forceinline__ v8f wmma16(v16h a, v16h b, v8f c) {
  return __builtin_amdgcn_wmma_f32_16x16x32_f16(false, a, false, b, (short)0, c,
                                                false, false);
}

// ---------------------------------------------------------------------------
// WMMA GEMM: out = act(A[MxK] * W[KxN] + bias) (+resid). W supplied TRANSPOSED
// (NxK row-major, row stride = K). Optional transposed f16 output for V.
// Block 128 = 4 waves (2x2), tile 64x64, K-step 32 via LDS.
// ---------------------------------------------------------------------------
__global__ __launch_bounds__(128) void gemm_kernel(
    const _Float16* __restrict__ A, int lda, const _Float16* __restrict__ Wt,
    const float* __restrict__ bias, const float* __restrict__ resid,
    float* __restrict__ outF32, _Float16* __restrict__ outF16,
    _Float16* __restrict__ outT, int tRows, int tLd, int M, int N, int K,
    int act) {
  __shared__ _Float16 As[64 * 48];  // 64 rows(m) x 32 halfs(k), stride 48
  __shared__ _Float16 Bs[64 * 48];  // 64 rows(n) x 32 halfs(k), stride 48
  const int mBase = blockIdx.y * 64;
  const int nBase = blockIdx.x * 64;
  const int wave = (int)(threadIdx.x >> 5);
  const unsigned lane = laneid();
  const int wr = wave >> 1, wc = wave & 1;
  v8f C00 = zero8(), C01 = zero8(), C10 = zero8(), C11 = zero8();

  const int t = (int)threadIdx.x;
  const int arow = t >> 1, ahalf = t & 1;  // 64 rows x 2 halves of 16 halfs
  int grow = mBase + arow;
  if (grow >= M) grow = M - 1;
  const int nrow = nBase + arow;  // weight rows never OOB (N multiple of 64)

  for (int k0 = 0; k0 < K; k0 += 32) {
    __syncthreads();
    {
      stage16(A + (size_t)grow * lda + k0 + ahalf * 16,
              &As[arow * 48 + ahalf * 16]);
      stage16(Wt + (size_t)nrow * K + k0 + ahalf * 16,
              &Bs[arow * 48 + ahalf * 16]);
      if (k0 + 32 < K) {
        __builtin_prefetch(A + (size_t)grow * lda + k0 + 32, 0, 3);
        __builtin_prefetch(Wt + (size_t)nrow * K + k0 + 32, 0, 3);
      }
    }
    async_fence();
    __syncthreads();
    v16h a0 = load_a_frag(&As[(wr * 32 + 0) * 48], 48);
    v16h a1 = load_a_frag(&As[(wr * 32 + 16) * 48], 48);
    v16h b0 = load_bt_frag(&Bs[(wc * 32 + 0) * 48], 48);
    v16h b1 = load_bt_frag(&Bs[(wc * 32 + 16) * 48], 48);
    C00 = wmma16(a0, b0, C00);
    C01 = wmma16(a0, b1, C01);
    C10 = wmma16(a1, b0, C10);
    C11 = wmma16(a1, b1, C11);
  }

#pragma unroll
  for (int i = 0; i < 2; ++i) {
#pragma unroll
    for (int j = 0; j < 2; ++j) {
      v8f Cv = (i == 0) ? (j == 0 ? C00 : C01) : (j == 0 ? C10 : C11);
#pragma unroll
      for (int r = 0; r < 8; ++r) {
        int m = mBase + wr * 32 + i * 16 + ((int)(lane >> 4)) * 8 + r;
        int n = nBase + wc * 32 + j * 16 + (int)(lane & 15u);
        if (m < M) {
          float vv = Cv[r] + bias[n];
          if (act == 1) vv = fmaxf(vv, 0.0f);
          else if (act == 2) vv = vv / (1.0f + __expf(-vv));
          if (resid) vv += resid[(size_t)m * N + n];
          if (outF32) outF32[(size_t)m * N + n] = vv;
          if (outF16) outF16[(size_t)m * N + n] = (_Float16)vv;
          if (outT) {
            int bb = m / tRows;
            int rr = m - bb * tRows;
            outT[((size_t)bb * N + n) * tLd + rr] = (_Float16)vv;
          }
        }
      }
    }
  }
}

// ---------------------------------------------------------------------------
// Flash attention (one head, d=64). Block 128 = 4 waves; wave owns 16 q-rows.
// K in row-major (keys x 64); V pre-transposed (channels x keys, stride tLdV).
// ---------------------------------------------------------------------------
__global__ __launch_bounds__(128) void flash_attn_kernel(
    const _Float16* __restrict__ Q, const _Float16* __restrict__ K,
    const _Float16* __restrict__ VT, _Float16* __restrict__ O,
    const unsigned char* __restrict__ qmask,
    const unsigned char* __restrict__ kmask, int nq, int nk, int qBatchRows,
    int kBatchRows, int tLdV, int qmStride, int kmStride, float scale) {
  __shared__ _Float16 Ks[32 * 72];   // keys x 64 halfs (stride 72)
  __shared__ _Float16 Vt[64 * 48];   // channels x 32 keys (stride 48)
  __shared__ float kmLds[32];
  __shared__ _Float16 Ps[4 * 16 * 36];

  const int b = (int)blockIdx.z, h = (int)blockIdx.y;
  const int wave = (int)(threadIdx.x >> 5);
  const unsigned lane = laneid();
  const int q0 = (int)blockIdx.x * 64 + wave * 16;

  // Q fragments (two 16x32 K-slices over d=0..63) in registers
  v16h aQ0, aQ1;
  {
    unsigned m = lane & 15u, kh = lane >> 4;
    int qrow = q0 + (int)m;
    if (qrow >= nq) qrow = nq - 1;
    const _Float16* qp = Q + (size_t)(b * qBatchRows + qrow) * DD + h * DSUB;
#pragma unroll
    for (int v = 0; v < 8; ++v) {
      int k = ((v >> 2) << 4) + ((int)kh << 3) + ((v & 3) << 1);
      aQ0[2 * v] = qp[k];      aQ0[2 * v + 1] = qp[k + 1];
      aQ1[2 * v] = qp[32 + k]; aQ1[2 * v + 1] = qp[32 + k + 1];
    }
  }
  bool qok[8];
#pragma unroll
  for (int r = 0; r < 8; ++r) {
    int q = q0 + ((int)(lane >> 4)) * 8 + r;
    qok[r] = (q < nq) && (qmask[(size_t)b * qmStride + (q < nq ? q : 0)] != 0);
  }
  float mrow[8], lrow[8];
  v8f Oacc[4];
#pragma unroll
  for (int r = 0; r < 8; ++r) { mrow[r] = -3.0e38f; lrow[r] = 0.0f; }
#pragma unroll
  for (int tt = 0; tt < 4; ++tt) Oacc[tt] = zero8();

  const int nkb = (nk + 31) >> 5;
  for (int kb = 0; kb < nkb; ++kb) {
    __syncthreads();
    {  // cooperative loads: K (keys-major), V^T (channel-major), key masks
      int t = (int)threadIdx.x;
      int row = t >> 2, q4 = t & 3;  // K tile: 32 rows x 4 quarters
      int krow = kb * 32 + row;
      if (krow < nk) {
        stage16(K + (size_t)(b * kBatchRows + krow) * DD + h * DSUB + q4 * 16,
                &Ks[row * 72 + q4 * 16]);
      } else {
        zfill16(&Ks[row * 72 + q4 * 16]);
      }

      int drow = t >> 1, dhalf = t & 1;  // V^T tile: 64 rows x 2 halves
      stage16(VT + ((size_t)(b * DD + h * DSUB + drow)) * tLdV + kb * 32 +
                  dhalf * 16,
              &Vt[drow * 48 + dhalf * 16]);

      if (t < 32) {
        int kk = kb * 32 + t;
        kmLds[t] = (kk < nk)
                       ? (kmask[(size_t)b * kmStride + kk] ? 1.0f : 0.0f)
                       : -1.0f;  // -1 marks padded key
      }
    }
    async_fence();
    __syncthreads();

    // S = Q * K^T : contiguous per-lane B fragments from Ks
    v8f S0 = zero8(), S1 = zero8();
    {
      v16h b00 = load_bt_frag(&Ks[0 * 72 + 0], 72);    // keys 0-15, d 0-31
      v16h b01 = load_bt_frag(&Ks[0 * 72 + 32], 72);   // keys 0-15, d 32-63
      v16h b10 = load_bt_frag(&Ks[16 * 72 + 0], 72);   // keys 16-31, d 0-31
      v16h b11 = load_bt_frag(&Ks[16 * 72 + 32], 72);  // keys 16-31, d 32-63
      S0 = wmma16(aQ0, b00, S0);
      S0 = wmma16(aQ1, b01, S0);
      S1 = wmma16(aQ0, b10, S1);
      S1 = wmma16(aQ1, b11, S1);
    }

    // online softmax; row reductions live inside 16-lane C-fragment halves
    float c0 = kmLds[lane & 15u];
    float c1 = kmLds[16 + (lane & 15u)];
#pragma unroll
    for (int r = 0; r < 8; ++r) {
      float s0 = S0[r], s1 = S1[r];
      s0 = (c0 < 0.0f) ? -1e30f
                       : ((qok[r] && c0 > 0.5f) ? s0 * scale : -1e9f);
      s1 = (c1 < 0.0f) ? -1e30f
                       : ((qok[r] && c1 > 0.5f) ? s1 * scale : -1e9f);
      float mn = fmaxf(s0, s1);
#pragma unroll
      for (int off = 1; off < 16; off <<= 1)
        mn = fmaxf(mn, __shfl_xor(mn, off, 32));
      float mc = fmaxf(mrow[r], mn);
      float fac = __expf(mrow[r] - mc);
      float p0 = __expf(s0 - mc);
      float p1 = __expf(s1 - mc);
      float rs = p0 + p1;
#pragma unroll
      for (int off = 1; off < 16; off <<= 1) rs += __shfl_xor(rs, off, 32);
      lrow[r] = lrow[r] * fac + rs;
      mrow[r] = mc;
#pragma unroll
      for (int tt = 0; tt < 4; ++tt) Oacc[tt][r] = Oacc[tt][r] * fac;
      int m = ((int)(lane >> 4)) * 8 + r;
      Ps[wave * 16 * 36 + m * 36 + (int)(lane & 15u)] = (_Float16)p0;
      Ps[wave * 16 * 36 + m * 36 + 16 + (int)(lane & 15u)] = (_Float16)p1;
    }
    __syncthreads();

    // O += P * V ; B fragments contiguous from channel-major Vt
    {
      v16h aP = load_a_frag(&Ps[wave * 16 * 36], 36);
#pragma unroll
      for (int vt = 0; vt < 4; ++vt) {
        v16h bV = load_bt_frag(&Vt[(vt * 16) * 48], 48);
        Oacc[vt] = wmma16(aP, bV, Oacc[vt]);
      }
    }
  }

#pragma unroll
  for (int r = 0; r < 8; ++r) {
    int q = q0 + ((int)(lane >> 4)) * 8 + r;
    if (q < nq) {
      float inv = 1.0f / lrow[r];
#pragma unroll
      for (int vt = 0; vt < 4; ++vt) {
        O[(size_t)(b * qBatchRows + q) * DD + h * DSUB + vt * 16 +
          (int)(lane & 15u)] = (_Float16)(Oacc[vt][r] * inv);
      }
    }
  }
}

// ---------------------------------------------------------------------------
// InstanceNorm over last dim (256), writes f16.
// ---------------------------------------------------------------------------
__global__ __launch_bounds__(256) void instnorm_kernel(
    const float* __restrict__ src, _Float16* __restrict__ dst) {
  __shared__ float redS[8], redS2[8];
  int row = (int)blockIdx.x;
  float x = src[(size_t)row * DD + threadIdx.x];
  float s = x, s2 = x * x;
#pragma unroll
  for (int off = 16; off; off >>= 1) {
    s += __shfl_xor(s, off, 32);
    s2 += __shfl_xor(s2, off, 32);
  }
  int wv = (int)(threadIdx.x >> 5);
  if ((threadIdx.x & 31u) == 0) { redS[wv] = s; redS2[wv] = s2; }
  __syncthreads();
  float ts = 0.0f, ts2 = 0.0f;
#pragma unroll
  for (int i = 0; i < 8; ++i) { ts += redS[i]; ts2 += redS2[i]; }
  float mean = ts * (1.0f / 256.0f);
  float var = ts2 * (1.0f / 256.0f) - mean * mean;
  float inv = rsqrtf(var + 1e-5f);
  dst[(size_t)row * DD + threadIdx.x] = (_Float16)((x - mean) * inv);
}

// ---------------------------------------------------------------------------
// Edge feature builder: sin/cos positional embed + semantics projection.
// ---------------------------------------------------------------------------
__global__ __launch_bounds__(256) void build_edges_kernel(
    const float* __restrict__ corners, const unsigned char* __restrict__ g,
    const float* __restrict__ sem, const float* __restrict__ rl,
    const float* __restrict__ W_sem, const float* __restrict__ b_sem,
    float* __restrict__ edges) {
  int e = (int)blockIdx.x, b = (int)blockIdx.y;
  int i = e / BN, j = e % BN;
  float pad = g[(size_t)b * EE + e] ? 1.0f : 0.0f;
  float lam = rl[(size_t)b * EE + e];
  float c1x = (corners[((size_t)b * BN + i) * 2 + 0] * 128.0f + 128.0f) * pad;
  float c1y = (corners[((size_t)b * BN + i) * 2 + 1] * 128.0f + 128.0f) * pad;
  float c2x = (corners[((size_t)b * BN + j) * 2 + 0] * 128.0f + 128.0f) * pad;
  float c2y = (corners[((size_t)b * BN + j) * 2 + 1] * 128.0f + 128.0f) * pad;
  float c3x = c1x * lam + c2x * (1.0f - lam);
  float c3y = c1y * lam + c2y * (1.0f - lam);
  int d = (int)threadIdx.x;
  float val;
  if (d < 128) {
    int coord = d >> 6;
    int w = d & 63;
    int tt = w >> 1;
    float dv = __expf(-9.210340371976184f * (float)tt * (1.0f / 32.0f));
    float a1 = (coord ? c1y : c1x) * dv;
    float a2 = (coord ? c2y : c2x) * dv;
    float a3 = (coord ? c3y : c3x) * dv;
    val = (w & 1) ? (__cosf(a1) + __cosf(a2) + __cosf(a3))
                  : (__sinf(a1) + __sinf(a2) + __sinf(a3));
  } else {
    int s = d - 128;
    float acc = 3.0f * b_sem[s];
#pragma unroll
    for (int k = 0; k < 7; ++k) {
      float es = (sem[((size_t)b * BN + i) * 7 + k] +
                  sem[((size_t)b * BN + j) * 7 + k]) * pad;
      acc += es * W_sem[k * 128 + s];
    }
    val = acc;
  }
  edges[((size_t)b * EE + e) * DD + d] = val;
}

// --------------------------- small utility kernels -------------------------
struct Ptr10 { const float* p[10]; };
struct Ptr3  { const float* p[3]; };

// fp32 -> f16 with transpose: dst holds W^T (N x K row-major).
__global__ void convert_layer_kernel(Ptr10 w, _Float16* dst) {
  const int sizes[10] = {262144, 262144, 65536, 65536, 65536,
                         65536,  65536,  65536, 65536, 65536};
  const int Kd[10] = {256, 1024, 256, 256, 256, 256, 256, 256, 256, 256};
  const int Nd[10] = {1024, 256, 256, 256, 256, 256, 256, 256, 256, 256};
  int total = 1048576;
  for (int idx = (int)(blockIdx.x * blockDim.x + threadIdx.x); idx < total;
       idx += (int)(gridDim.x * blockDim.x)) {
    int off = idx, m = 0;
    while (off >= sizes[m]) { off -= sizes[m]; ++m; }
    int n = off / Kd[m];
    int k = off - n * Kd[m];
    dst[idx] = (_Float16)w.p[m][k * Nd[m] + n];
  }
}

__global__ void convert_misc_kernel(Ptr3 w, _Float16* lw, _Float16* bb) {
  int total = 196608;
  for (int idx = (int)(blockIdx.x * blockDim.x + threadIdx.x); idx < total;
       idx += (int)(gridDim.x * blockDim.x)) {
    if (idx < 131072) {  // lW1, lW2: 256x256 transposed
      int mi = idx >> 16;
      int off = idx & 65535;
      int n = off >> 8, k = off & 255;
      lw[idx] = (_Float16)w.p[mi][k * 256 + n];
    } else {
      bb[idx - 131072] = (_Float16)w.p[2][idx - 131072];  // activations: as-is
    }
  }
}

__global__ void zero16_kernel(_Float16* __restrict__ dst, int n) {
  for (int idx = (int)(blockIdx.x * blockDim.x + threadIdx.x); idx < n;
       idx += (int)(gridDim.x * blockDim.x))
    dst[idx] = (_Float16)0.0f;
}

__global__ void f2h_kernel(const float* __restrict__ src,
                           _Float16* __restrict__ dst, int n) {
  for (int idx = (int)(blockIdx.x * blockDim.x + threadIdx.x); idx < n;
       idx += (int)(gridDim.x * blockDim.x))
    dst[idx] = (_Float16)src[idx];
}

__global__ void copy_kernel(const float* __restrict__ src,
                            float* __restrict__ dst, int n) {
  int idx = (int)(blockIdx.x * blockDim.x + threadIdx.x);
  if (idx < n) dst[idx] = src[idx];
}

__global__ void head_edges_kernel(const float* __restrict__ edges,
                                  const float* __restrict__ W,
                                  const float* __restrict__ b,
                                  float* __restrict__ out) {
  int idx = (int)(blockIdx.x * blockDim.x + threadIdx.x);
  if (idx >= MEROWS * 2) return;
  int m = idx >> 1, c = idx & 1;
  float acc = b[c];
  const float* e = edges + (size_t)m * DD;
#pragma unroll 8
  for (int d = 0; d < DD; ++d) acc += e[d] * W[d * 2 + c];
  out[idx] = acc;
}

__global__ void head_lambda_kernel(const _Float16* __restrict__ h2,
                                   const float* __restrict__ W,
                                   const float* __restrict__ b,
                                   float* __restrict__ out) {
  int m = (int)(blockIdx.x * blockDim.x + threadIdx.x);
  if (m >= MEROWS) return;
  float acc = b[0];
  const _Float16* hp = h2 + (size_t)m * DD;
#pragma unroll 8
  for (int d = 0; d < DD; ++d) acc += (float)hp[d] * W[d];
  out[m] = acc;
}

// ---------------------------------------------------------------------------
extern "C" void kernel_launch(void* const* d_in, const int* in_sizes, int n_in,
                              void* d_out, int out_size, void* d_ws,
                              size_t ws_size, hipStream_t stream) {
  (void)in_sizes; (void)n_in; (void)out_size; (void)ws_size;
  const float* corners = (const float*)d_in[0];
  const unsigned char* gmask = (const unsigned char*)d_in[1];
  // d_in[2] = corners_padding_mask (unused by reference)
  const float* semantics = (const float*)d_in[3];
  const float* bbemb = (const float*)d_in[4];
  const unsigned char* bbmask = (const unsigned char*)d_in[5];
  const float* rlam = (const float*)d_in[6];
  auto F = [&](int i) { return (const float*)d_in[i]; };
  // params flattened in JAX sorted-key pytree order
  const int PB = 7;
  const float* W_edges = F(PB + 0);
  const float* W_sem   = F(PB + 1);
  const float* b_edges = F(PB + 2);
  const float* b_sem   = F(PB + 3);
  const float* lW1 = F(PB + 4);
  const float* lW2 = F(PB + 5);
  const float* lW3 = F(PB + 6);
  const int LAY0 = PB + 7;  // 12 layers x 20 tensors
  const float* lb1 = F(LAY0 + 12 * 20 + 0);
  const float* lb2 = F(LAY0 + 12 * 20 + 1);
  const float* lb3 = F(LAY0 + 12 * 20 + 2);

  // workspace layout (bytes; total ~84.8 MB)
  char* ws = (char*)d_ws;
  float*    edges = (float*)(ws + 0);                 // 11,505,664
  _Float16* tmp16 = (_Float16*)(ws + 11505664);       //  5,752,832
  _Float16* q16   = (_Float16*)(ws + 17258496);       //  5,752,832
  _Float16* k16   = (_Float16*)(ws + 23011328);       //  5,752,832
  _Float16* vt16  = (_Float16*)(ws + 28764160);       //  5,767,168 (B*256*VTLD)
  _Float16* ao16  = (_Float16*)(ws + 34531328);       //  5,752,832
  _Float16* hid16 = (_Float16*)(ws + 40284160);       // 23,011,328
  _Float16* w16   = (_Float16*)(ws + 63295488);       // 25,165,824
  _Float16* lw16  = (_Float16*)(ws + 88461312);       //    262,144
  _Float16* bb16  = (_Float16*)(ws + 88723456);       //    131,072

  // prep: transpose+convert weights, zero V^T padding region
  for (int l = 0; l < 12; ++l) {
    Ptr10 p;
    int base = LAY0 + l * 20;
    p.p[0] = F(base + 0);   // W1
    p.p[1] = F(base + 1);   // W2
    p.p[2] = F(base + 4);   // ca.WK
    p.p[3] = F(base + 5);   // ca.WQ
    p.p[4] = F(base + 6);   // ca.WV
    p.p[5] = F(base + 7);   // ca.Wf
    p.p[6] = F(base + 12);  // ga.WK
    p.p[7] = F(base + 13);  // ga.WQ
    p.p[8] = F(base + 14);  // ga.WV
    p.p[9] = F(base + 15);  // ga.Wf
    convert_layer_kernel<<<1024, 256, 0, stream>>>(p, w16 + (size_t)l * 1048576);
  }
  {
    Ptr3 p; p.p[0] = lW1; p.p[1] = lW2; p.p[2] = bbemb;
    convert_misc_kernel<<<768, 256, 0, stream>>>(p, lw16, bb16);
  }
  zero16_kernel<<<2048, 256, 0, stream>>>(vt16, BBATCH * DD * VTLD);

  build_edges_kernel<<<dim3(EE, BBATCH), 256, 0, stream>>>(
      corners, gmask, semantics, rlam, W_sem, b_sem, edges);

  const int GM = (MEROWS + 63) / 64;  // 176 M-tiles
  for (int l = 0; l < 12; ++l) {
    int base = LAY0 + l * 20;
    const _Float16* lw = w16 + (size_t)l * 1048576;
    const _Float16* W1h = lw + 0;
    const _Float16* W2h = lw + 262144;
    const _Float16* caWK = lw + 524288;
    const _Float16* caWQ = lw + 589824;
    const _Float16* caWV = lw + 655360;
    const _Float16* caWf = lw + 720896;
    const _Float16* gaWK = lw + 786432;
    const _Float16* gaWQ = lw + 851968;
    const _Float16* gaWV = lw + 917504;
    const _Float16* gaWf = lw + 983040;
    const float* b1 = F(base + 2);
    const float* b2 = F(base + 3);
    const float* ca_bK = F(base + 8),  *ca_bQ = F(base + 9);
    const float* ca_bV = F(base + 10), *ca_bf = F(base + 11);
    const float* ga_bK = F(base + 16), *ga_bQ = F(base + 17);
    const float* ga_bV = F(base + 18), *ga_bf = F(base + 19);

    // ----- self attention -----
    instnorm_kernel<<<MEROWS, 256, 0, stream>>>(edges, tmp16);
    gemm_kernel<<<dim3(4, GM), 128, 0, stream>>>(tmp16, 256, gaWQ, ga_bQ,
        nullptr, nullptr, q16, nullptr, 0, 0, MEROWS, 256, 256, 0);
    gemm_kernel<<<dim3(4, GM), 128, 0, stream>>>(tmp16, 256, gaWK, ga_bK,
        nullptr, nullptr, k16, nullptr, 0, 0, MEROWS, 256, 256, 0);
    gemm_kernel<<<dim3(4, GM), 128, 0, stream>>>(tmp16, 256, gaWV, ga_bV,
        nullptr, nullptr, nullptr, vt16, EE, VTLD, MEROWS, 256, 256, 0);
    flash_attn_kernel<<<dim3((EE + 63) / 64, HH, BBATCH), 128, 0, stream>>>(
        q16, k16, vt16, ao16, gmask, gmask, EE, EE, EE, EE, VTLD, EE, EE,
        0.125f);
    gemm_kernel<<<dim3(4, GM), 128, 0, stream>>>(ao16, 256, gaWf, ga_bf,
        edges, edges, nullptr, nullptr, 0, 0, MEROWS, 256, 256, 0);

    // ----- cross attention -----
    instnorm_kernel<<<MEROWS, 256, 0, stream>>>(edges, tmp16);
    gemm_kernel<<<dim3(4, GM), 128, 0, stream>>>(tmp16, 256, caWQ, ca_bQ,
        nullptr, nullptr, q16, nullptr, 0, 0, MEROWS, 256, 256, 0);
    gemm_kernel<<<dim3(4, 4), 128, 0, stream>>>(bb16, 256, caWK, ca_bK,
        nullptr, nullptr, k16, nullptr, 0, 0, BBATCH * LBB, 256, 256, 0);
    gemm_kernel<<<dim3(4, 4), 128, 0, stream>>>(bb16, 256, caWV, ca_bV,
        nullptr, nullptr, nullptr, vt16, LBB, LBB, BBATCH * LBB, 256, 256, 0);
    flash_attn_kernel<<<dim3((EE + 63) / 64, HH, BBATCH), 128, 0, stream>>>(
        q16, k16, vt16, ao16, gmask, bbmask, EE, LBB, EE, LBB, LBB, EE, LBB,
        0.125f);
    gemm_kernel<<<dim3(4, GM), 128, 0, stream>>>(ao16, 256, caWf, ca_bf,
        edges, edges, nullptr, nullptr, 0, 0, MEROWS, 256, 256, 0);

    // ----- FFN -----
    instnorm_kernel<<<MEROWS, 256, 0, stream>>>(edges, tmp16);
    gemm_kernel<<<dim3(16, GM), 128, 0, stream>>>(tmp16, 256, W1h, b1,
        nullptr, nullptr, hid16, nullptr, 0, 0, MEROWS, 1024, 256, 1);
    gemm_kernel<<<dim3(4, GM), 128, 0, stream>>>(hid16, 1024, W2h, b2,
        edges, edges, nullptr, nullptr, 0, 0, MEROWS, 256, 1024, 0);
  }

  // heads: out_edges | random_lambda | out_lambdas
  float* out = (float*)d_out;
  head_edges_kernel<<<(MEROWS * 2 + 255) / 256, 256, 0, stream>>>(
      edges, W_edges, b_edges, out);
  copy_kernel<<<(MEROWS + 255) / 256, 256, 0, stream>>>(rlam, out + MEROWS * 2,
                                                        MEROWS);
  f2h_kernel<<<4096, 256, 0, stream>>>(edges, tmp16, MEROWS * DD);
  gemm_kernel<<<dim3(4, GM), 128, 0, stream>>>(tmp16, 256, lw16, lb1,
      nullptr, nullptr, q16, nullptr, 0, 0, MEROWS, 256, 256, 2);
  gemm_kernel<<<dim3(4, GM), 128, 0, stream>>>(q16, 256, lw16 + 65536, lb2,
      nullptr, nullptr, k16, nullptr, 0, 0, MEROWS, 256, 256, 2);
  head_lambda_kernel<<<(MEROWS + 255) / 256, 256, 0, stream>>>(
      k16, lW3, lb3, out + MEROWS * 3);
}
